// PPAPLayer_74045236183648
// MI455X (gfx1250) — matrix operand order repaired and checked by reading, not compiled, not run on hardware
//
#include <hip/hip_runtime.h>

// ---------------- problem constants ----------------
#define NNODES   20000
#define DNODE    1280
#define NEDGES   100000
#define DEDGE    505
#define DEDGEP   512           // edge_core row padded to 512 bf16
#define DCAT     3065
#define KPAD     3072          // 1280 | 512 | 1280 segment-padded K
#define NHEADS   20
#define ATTEMB   1280
#define HIDDEN   5120
#define KSTEPS   96            // KPAD / 32

// output layout (floats): h | edge_core | global_fea | att
#define H_ELEMS   25600000
#define EC_ELEMS  50500000
#define GF_OFF    76100000
#define ATT_OFF   76101280

// workspace layout (bytes); total ~170 MB
#define WS_HB      0            // 20000*1280 bf16 = 51,200,000
#define WS_ECB     51200000     // 100000*512 bf16 = 102,400,000
#define WS_WTATT   153600000    // 32*3072 bf16
#define WS_WTNODE  153796608    // 1280*3072 bf16
#define WS_WBUF    161660928    // 100000*20 f32
#define WS_SMALL   169660928
#define WS_HMAX    (WS_SMALL)
#define WS_HSUM    (WS_SMALL + 128)
#define WS_GF      (WS_SMALL + 256)
#define WS_T1      (WS_SMALL + 5376)
#define WS_T2      (WS_SMALL + 25856)

// ---------------- types ----------------
typedef __attribute__((ext_vector_type(16))) __bf16 v16bf;
typedef __attribute__((ext_vector_type(8)))  float  v8f;
typedef unsigned int u32x4 __attribute__((ext_vector_type(4)));
typedef int          i32x4 __attribute__((ext_vector_type(4)));
typedef int          i32x8 __attribute__((ext_vector_type(8)));

struct __align__(16) U4 { unsigned int x, y, z, w; };
union FragBF { v16bf v; U4 q[2]; };

#if defined(__AMDGCN__) && __has_builtin(__builtin_amdgcn_tensor_load_to_lds)
#define HAVE_TDM 1
#else
#define HAVE_TDM 0
#endif

#if HAVE_TDM
#if __has_include(<hip/amd_detail/amd_gfx1250_TDM.h>)
#define TDM_CALL(g0, g1, g2, g3) \
  __builtin_amdgcn_tensor_load_to_lds(g0, g1, g2, g3, (i32x8){0,0,0,0,0,0,0,0}, 0)
#else
#define TDM_CALL(g0, g1, g2, g3) \
  __builtin_amdgcn_tensor_load_to_lds(g0, g1, g2, g3, 0)
#endif

// Fill D# group1 for a 2D bf16 tensor: width elems/row, dim0 stride == width.
static __device__ __forceinline__ i32x8 tdm_group1(int width, int tensor_rows, int tile_rows) {
  i32x8 g1;
  g1[0] = 0x00010000;                                       // data_size = 1 -> 2 bytes
  g1[1] = (int)(((unsigned)width & 0xffffu) << 16);         // tensor_dim0 lo16
  g1[2] = (int)((((unsigned)width >> 16) & 0xffffu) |
                (((unsigned)tensor_rows & 0xffffu) << 16)); // dim0 hi16 | dim1 lo16
  g1[3] = (int)((((unsigned)tensor_rows >> 16) & 0xffffu) |
                (((unsigned)width & 0xffffu) << 16));       // dim1 hi16 | tile_dim0
  g1[4] = tile_rows;                                        // tile_dim1 (tile_dim2 = 0)
  g1[5] = width;                                            // tensor_dim0_stride lo32
  g1[6] = 0;
  g1[7] = 0;
  return g1;
}

static __device__ __forceinline__ u32x4 tdm_group0(unsigned w0, unsigned ldsAddr,
                                                   const void* gaddr) {
  unsigned long long ga = (unsigned long long)(uintptr_t)gaddr;
  u32x4 g0;
  g0[0] = w0;                                               // count=1 (+gather bits)
  g0[1] = ldsAddr;                                          // LDS byte address
  g0[2] = (unsigned)(ga & 0xffffffffu);
  g0[3] = (unsigned)((ga >> 32) & 0x01ffffffu) | 0x80000000u;  // type = 2 ("image")
  return g0;
}

// Gather 16 rows (16-bit row indices) of a [tensor_rows x width] bf16 tensor into LDS.
static __device__ __forceinline__ void tdm_gather16(unsigned ldsAddr,
                                                    const unsigned short* base,
                                                    int width, int tensor_rows,
                                                    const int* __restrict__ rowIdx) {
  u32x4 g0 = tdm_group0(0x80000001u, ldsAddr, base);        // gather_mode=1, 16-bit idx
  i32x8 g1 = tdm_group1(width, tensor_rows, 16);            // tile_dim1 = #indices
  i32x4 g2, g3;
  #pragma unroll
  for (int j = 0; j < 4; ++j) {
    g2[j] = (int)(((unsigned)rowIdx[2*j]   & 0xffffu) |
                  (((unsigned)rowIdx[2*j+1] & 0xffffu) << 16));
    g3[j] = (int)(((unsigned)rowIdx[8+2*j]   & 0xffffu) |
                  (((unsigned)rowIdx[8+2*j+1] & 0xffffu) << 16));
  }
  TDM_CALL(g0, g1, g2, g3);
}

// Plain 2D tile load: tile_rows consecutive rows starting at tileBase.
static __device__ __forceinline__ void tdm_load2d(unsigned ldsAddr,
                                                  const unsigned short* tileBase,
                                                  int width, int tensor_rows, int tile_rows) {
  u32x4 g0 = tdm_group0(0x00000001u, ldsAddr, tileBase);
  i32x8 g1 = tdm_group1(width, tensor_rows, tile_rows);
  i32x4 z = {0, 0, 0, 0};
  TDM_CALL(g0, g1, z, z);
}

static __device__ __forceinline__ void tdm_wait0() {
#if __has_builtin(__builtin_amdgcn_s_wait_tensorcnt)
  __builtin_amdgcn_s_wait_tensorcnt(0);
#endif
}
#endif // HAVE_TDM

static __device__ __forceinline__ unsigned short f2bf(float f) {
  unsigned int u = __float_as_uint(f);
  u += 0x7FFFu + ((u >> 16) & 1u);
  return (unsigned short)(u >> 16);
}

static __device__ __forceinline__ float gelu_erf(float x) {
  return 0.5f * x * (1.0f + erff(x * 0.70710678118654752440f));
}

// ---------------- init ----------------
__global__ __launch_bounds__(256) void init_kernel(unsigned int* __restrict__ hmax,
                                                   float* __restrict__ hsum,
                                                   float* __restrict__ gf) {
  int i = blockIdx.x * 256 + threadIdx.x;
  if (i < NHEADS) { hmax[i] = 0u; hsum[i] = 0.0f; }
  if (i < ATTEMB) gf[i] = 0.0f;
}

// ---------------- f32 -> bf16 conversions ----------------
__global__ __launch_bounds__(256) void convert_h_kernel(const float* __restrict__ h,
                                                        unsigned short* __restrict__ hb) {
  long i = (long)blockIdx.x * 256 + threadIdx.x;
  if (i < (long)NNODES * DNODE) hb[i] = f2bf(h[i]);
}

__global__ __launch_bounds__(256) void convert_ec_kernel(const float* __restrict__ ec,
                                                         unsigned short* __restrict__ ecb) {
  long i = (long)blockIdx.x * 256 + threadIdx.x;
  if (i < (long)NEDGES * DEDGEP) {
    int r = (int)(i >> 9), c = (int)(i & (DEDGEP - 1));
    ecb[i] = (c < DEDGE) ? f2bf(ec[(long)r * DEDGE + c]) : (unsigned short)0;
  }
}

// map segment-padded K index -> original cat K (or -1 for pad)
static __device__ __forceinline__ int ksrc(int kp) {
  if (kp < DNODE) return kp;
  if (kp < DNODE + DEDGEP) { int c = kp - DNODE; return (c < DEDGE) ? DNODE + c : -1; }
  return DCAT - DNODE + (kp - DNODE - DEDGEP) + DNODE;   // 3065 + (kp - 1792)
}

__global__ __launch_bounds__(256) void convert_w_kernel(const float* __restrict__ Wa,
                                                        const float* __restrict__ Wn,
                                                        unsigned short* __restrict__ Ta,
                                                        unsigned short* __restrict__ Tn) {
  long i = (long)blockIdx.x * 256 + threadIdx.x;
  const long TOTA = 32L * KPAD;
  const long TOTN = (long)ATTEMB * KPAD;
  if (i < TOTA) {
    int n = (int)(i / KPAD), kp = (int)(i % KPAD);
    int k = ksrc(kp);
    float v = (n < NHEADS && k >= 0) ? Wa[(long)k * NHEADS + n] : 0.0f;
    Ta[i] = f2bf(v);
  } else if (i < TOTA + TOTN) {
    long j = i - TOTA;
    int n = (int)(j / KPAD), kp = (int)(j % KPAD);
    int k = ksrc(kp);
    float v = (k >= 0) ? Wn[(long)k * ATTEMB + n] : 0.0f;
    Tn[j] = f2bf(v);
  }
}

// select LDS region for a 32-wide K window (segments are 32-aligned)
static __device__ __forceinline__ const unsigned short*
seg_ptr(const unsigned short* hiS, const unsigned short* ecS,
        const unsigned short* hjS, int k, int* rstride, int* kl) {
  if (k < DNODE)          { *rstride = DNODE;  *kl = k;                 return hiS; }
  if (k < DNODE + DEDGEP) { *rstride = DEDGEP; *kl = k - DNODE;         return ecS; }
  *rstride = DNODE; *kl = k - DNODE - DEDGEP;                           return hjS;
}

// ---------------- attention logits (M=16, WMMA, TDM-staged) ----------------
__global__ __launch_bounds__(64) void logits_kernel(const unsigned short* __restrict__ hb,
                                                    const unsigned short* __restrict__ ecb,
                                                    const int* __restrict__ idx,
                                                    const unsigned short* __restrict__ Wt,  // [32][KPAD]
                                                    const float* __restrict__ b_att,
                                                    float* __restrict__ wbuf,
                                                    unsigned int* __restrict__ hmax) {
  extern __shared__ __align__(16) unsigned short dynS[];
  unsigned short* hiS = dynS;                 // 16*1280
  unsigned short* ecS = dynS + 16 * DNODE;    // 16*512
  unsigned short* hjS = ecS + 16 * DEDGEP;    // 16*1280

  const int tid  = threadIdx.x;
  const int e0   = blockIdx.x * 16;
  const int lane = tid & 31;
  const int wid  = tid >> 5;
  const int row  = lane & 15;
  const int s8   = (lane >> 4) * 8;

#if HAVE_TDM
  if (wid == 0) {
    tdm_gather16((unsigned)(uintptr_t)hiS, hb, DNODE, NNODES, idx + e0);
    tdm_load2d((unsigned)(uintptr_t)ecS, ecb + (long)e0 * DEDGEP, DEDGEP, NEDGES, 16);
    tdm_wait0();
  } else {
    tdm_gather16((unsigned)(uintptr_t)hjS, hb, DNODE, NNODES, idx + NEDGES + e0);
    tdm_wait0();
  }
  __syncthreads();
#else
  for (int u = tid; u < 16 * DNODE / 8; u += 64) {
    int r = u / 160, c = (u % 160) * 8;
    *(U4*)&hiS[r * DNODE + c] = *(const U4*)&hb[(long)idx[e0 + r] * DNODE + c];
    *(U4*)&hjS[r * DNODE + c] = *(const U4*)&hb[(long)idx[NEDGES + e0 + r] * DNODE + c];
  }
  for (int u = tid; u < 16 * DEDGEP / 8; u += 64) {
    int r = u / 64, c = (u % 64) * 8;
    *(U4*)&ecS[r * DEDGEP + c] = *(const U4*)&ecb[(long)(e0 + r) * DEDGEP + c];
  }
  __syncthreads();
#endif

  v8f acc = {0.f,0.f,0.f,0.f,0.f,0.f,0.f,0.f};
  const int n = wid * 16 + row;

  #pragma unroll 1
  for (int ks = 0; ks < KSTEPS; ++ks) {
    const int k = ks * 32;
    int rstride, kl;
    const unsigned short* base = seg_ptr(hiS, ecS, hjS, k, &rstride, &kl);
    FragBF a;
    const unsigned short* ap = base + row * rstride + kl + s8;
    a.q[0] = *reinterpret_cast<const U4*>(ap);
    a.q[1] = *reinterpret_cast<const U4*>(ap + 16);
    FragBF b;
    const unsigned short* bp = &Wt[(long)n * KPAD + k + s8];
    b.q[0] = *reinterpret_cast<const U4*>(bp);
    b.q[1] = *reinterpret_cast<const U4*>(bp + 16);
    acc = __builtin_amdgcn_wmma_f32_16x16x32_bf16(false, a.v, false, b.v,
                                                  (short)0, acc, false, false);
  }

  if (n < NHEADS) {
    const float ba = b_att[n];
    float localmax = 0.0f;
    #pragma unroll
    for (int v = 0; v < 8; ++v) {
      int m = s8 + v;                       // C layout: M = v (+8 upper half-lanes)
      float x = acc[v] + ba;
      x = (x > 0.0f ? x : 0.0f) * 0.125f;   // relu / sqrt(64)
      wbuf[(long)(e0 + m) * NHEADS + n] = x;
      localmax = fmaxf(localmax, x);
    }
    atomicMax(&hmax[n], __float_as_uint(localmax));
  }
}

// ---------------- softmax passes ----------------
__global__ __launch_bounds__(256) void expsum_kernel(float* __restrict__ wbuf,
                                                     const unsigned int* __restrict__ hmax,
                                                     float* __restrict__ hsum) {
  __shared__ float hs[NHEADS];
  const int tid = threadIdx.x;
  const int e   = blockIdx.x * 256 + tid;
  if (tid < NHEADS) hs[tid] = 0.0f;
  __syncthreads();
  if (e < NEDGES) {
    #pragma unroll
    for (int hh = 0; hh < NHEADS; ++hh) {
      float mx = __uint_as_float(hmax[hh]);
      float p  = expf(wbuf[(long)e * NHEADS + hh] - mx);
      wbuf[(long)e * NHEADS + hh] = p;
      atomicAdd(&hs[hh], p);
    }
  }
  __syncthreads();
  if (tid < NHEADS) atomicAdd(&hsum[tid], hs[tid]);
}

__global__ __launch_bounds__(256) void attwrite_kernel(const float* __restrict__ wbuf,
                                                       const float* __restrict__ hsum,
                                                       float* __restrict__ att_out) {
  long i = (long)blockIdx.x * 256 + threadIdx.x;
  if (i < (long)NEDGES * NHEADS) att_out[i] = wbuf[i] / hsum[(int)(i % NHEADS)];
}

// ------- value GEMM (M=32, bf16 WMMA) + GELU + attention pooling -------
__global__ __launch_bounds__(256) void gemm_pool_kernel(const unsigned short* __restrict__ hb,
                                                        const unsigned short* __restrict__ ecb,
                                                        const int* __restrict__ idx,
                                                        const unsigned short* __restrict__ Wt, // [1280][KPAD]
                                                        const float* __restrict__ b_node,
                                                        const float* __restrict__ att,
                                                        float* __restrict__ gf) {
  extern __shared__ __align__(16) unsigned short dynS[];
  unsigned short* hiS = dynS;                 // 32*1280
  unsigned short* ecS = dynS + 32 * DNODE;    // 32*512
  unsigned short* hjS = ecS + 32 * DEDGEP;    // 32*1280
  __shared__ float gfS[ATTEMB];
  __shared__ float attS[32 * NHEADS];

  const int tid  = threadIdx.x;
  const int e0   = blockIdx.x * 32;
  const int lane = tid & 31;
  const int wid  = tid >> 5;
  const int row  = lane & 15;
  const int s8   = (lane >> 4) * 8;

  for (int i = tid; i < ATTEMB; i += 256) gfS[i] = 0.0f;
  for (int i = tid; i < 32 * NHEADS; i += 256) attS[i] = att[(long)e0 * NHEADS + i];

#if HAVE_TDM
  if      (wid == 0) { tdm_gather16((unsigned)(uintptr_t)hiS,                 hb, DNODE, NNODES, idx + e0);               tdm_wait0(); }
  else if (wid == 1) { tdm_gather16((unsigned)(uintptr_t)(hiS + 16 * DNODE),  hb, DNODE, NNODES, idx + e0 + 16);          tdm_wait0(); }
  else if (wid == 2) { tdm_gather16((unsigned)(uintptr_t)hjS,                 hb, DNODE, NNODES, idx + NEDGES + e0);      tdm_wait0(); }
  else if (wid == 3) { tdm_gather16((unsigned)(uintptr_t)(hjS + 16 * DNODE),  hb, DNODE, NNODES, idx + NEDGES + e0 + 16); tdm_wait0(); }
  else if (wid == 4) { tdm_load2d((unsigned)(uintptr_t)ecS, ecb + (long)e0 * DEDGEP, DEDGEP, NEDGES, 32);                 tdm_wait0(); }
  __syncthreads();
#else
  for (int u = tid; u < 32 * DNODE / 8; u += 256) {
    int r = u / 160, c = (u % 160) * 8;
    *(U4*)&hiS[r * DNODE + c] = *(const U4*)&hb[(long)idx[e0 + r] * DNODE + c];
    *(U4*)&hjS[r * DNODE + c] = *(const U4*)&hb[(long)idx[NEDGES + e0 + r] * DNODE + c];
  }
  for (int u = tid; u < 32 * DEDGEP / 8; u += 256) {
    int r = u / 64, c = (u % 64) * 8;
    *(U4*)&ecS[r * DEDGEP + c] = *(const U4*)&ecb[(long)(e0 + r) * DEDGEP + c];
  }
  __syncthreads();
#endif

  v8f acc[2][10];
  #pragma unroll
  for (int mt = 0; mt < 2; ++mt)
    #pragma unroll
    for (int nt = 0; nt < 10; ++nt)
      acc[mt][nt] = (v8f){0.f,0.f,0.f,0.f,0.f,0.f,0.f,0.f};

  #pragma unroll 1
  for (int ks = 0; ks < KSTEPS; ++ks) {
    const int k = ks * 32;
    int rstride, kl;
    const unsigned short* base = seg_ptr(hiS, ecS, hjS, k, &rstride, &kl);
    FragBF a0, a1;
    const unsigned short* ap0 = base + row * rstride + kl + s8;
    const unsigned short* ap1 = base + (16 + row) * rstride + kl + s8;
    a0.q[0] = *reinterpret_cast<const U4*>(ap0);
    a0.q[1] = *reinterpret_cast<const U4*>(ap0 + 16);
    a1.q[0] = *reinterpret_cast<const U4*>(ap1);
    a1.q[1] = *reinterpret_cast<const U4*>(ap1 + 16);
    const int kg = k + s8;                      // B: K 8h..8h+7 and 16+8h..16+8h+7 per half-lane
    #pragma unroll
    for (int nt = 0; nt < 10; ++nt) {
      const int n = wid * 160 + nt * 16 + row;
      const unsigned short* bp = &Wt[(long)n * KPAD + kg];
      FragBF b;
      b.q[0] = *reinterpret_cast<const U4*>(bp);
      b.q[1] = *reinterpret_cast<const U4*>(bp + 16);
      acc[0][nt] = __builtin_amdgcn_wmma_f32_16x16x32_bf16(false, a0.v, false, b.v,
                                                           (short)0, acc[0][nt], false, false);
      acc[1][nt] = __builtin_amdgcn_wmma_f32_16x16x32_bf16(false, a1.v, false, b.v,
                                                           (short)0, acc[1][nt], false, false);
    }
  }
  __syncthreads();

  #pragma unroll
  for (int nt = 0; nt < 10; ++nt) {
    const int n    = wid * 160 + nt * 16 + row;
    const int head = n >> 6;
    const float bn = b_node[n];
    float partial = 0.0f;
    #pragma unroll
    for (int mt = 0; mt < 2; ++mt)
      #pragma unroll
      for (int v = 0; v < 8; ++v) {
        int m = mt * 16 + s8 + v;
        float g = gelu_erf(acc[mt][nt][v] + bn);
        partial += attS[m * NHEADS + head] * g;
      }
    atomicAdd(&gfS[n], partial);
  }
  __syncthreads();
  for (int i = tid; i < ATTEMB; i += 256) atomicAdd(&gf[i], gfS[i]);
}

// ---------------- FFN GEMVs ----------------
__global__ __launch_bounds__(256) void gemv_kernel(const float* __restrict__ x,
                                                   const float* __restrict__ W,
                                                   const float* __restrict__ b,
                                                   float* __restrict__ y,
                                                   int n_in, int n_out, int do_relu) {
  __shared__ float xs[HIDDEN];
  for (int i = threadIdx.x; i < n_in; i += 256) xs[i] = x[i];
  __syncthreads();
  int j = blockIdx.x * 256 + threadIdx.x;
  if (j < n_out) {
    float s = (b != nullptr) ? b[j] : 0.0f;
    for (int i = 0; i < n_in; ++i) s += xs[i] * W[(long)i * n_out + j];
    y[j] = do_relu ? fmaxf(s, 0.0f) : s;
  }
}

// ---------------- host launch sequence ----------------
extern "C" void kernel_launch(void* const* d_in, const int* in_sizes, int n_in,
                              void* d_out, int out_size, void* d_ws, size_t ws_size,
                              hipStream_t stream) {
  const float* h      = (const float*)d_in[0];
  const float* ec     = (const float*)d_in[1];
  const int*   idx    = (const int*)  d_in[2];
  const float* W_att  = (const float*)d_in[3];
  const float* b_att  = (const float*)d_in[4];
  const float* W_node = (const float*)d_in[5];
  const float* b_node = (const float*)d_in[6];
  const float* W1     = (const float*)d_in[7];
  const float* b1     = (const float*)d_in[8];
  const float* W2     = (const float*)d_in[9];
  const float* b2     = (const float*)d_in[10];
  const float* W_h    = (const float*)d_in[11];

  float* out = (float*)d_out;
  char*  ws  = (char*)d_ws;

  unsigned short* hb      = (unsigned short*)(ws + WS_HB);
  unsigned short* ecb     = (unsigned short*)(ws + WS_ECB);
  unsigned short* Wt_att  = (unsigned short*)(ws + WS_WTATT);
  unsigned short* Wt_node = (unsigned short*)(ws + WS_WTNODE);
  float*          wbuf    = (float*)(ws + WS_WBUF);
  unsigned int*   hmax    = (unsigned int*)(ws + WS_HMAX);
  float*          hsum    = (float*)(ws + WS_HSUM);
  float*          gf      = (float*)(ws + WS_GF);
  float*          t1      = (float*)(ws + WS_T1);
  float*          t2      = (float*)(ws + WS_T2);

  hipMemcpyAsync(out,           h,  (size_t)H_ELEMS  * sizeof(float),
                 hipMemcpyDeviceToDevice, stream);
  hipMemcpyAsync(out + H_ELEMS, ec, (size_t)EC_ELEMS * sizeof(float),
                 hipMemcpyDeviceToDevice, stream);

  init_kernel<<<(ATTEMB + 255) / 256, 256, 0, stream>>>(hmax, hsum, gf);

  convert_h_kernel<<<(unsigned)(((long)NNODES * DNODE + 255) / 256), 256, 0, stream>>>(h, hb);
  convert_ec_kernel<<<(unsigned)(((long)NEDGES * DEDGEP + 255) / 256), 256, 0, stream>>>(ec, ecb);
  {
    long total = 32L * KPAD + (long)ATTEMB * KPAD;
    convert_w_kernel<<<(unsigned)((total + 255) / 256), 256, 0, stream>>>(
        W_att, W_node, Wt_att, Wt_node);
  }

  // dynamic LDS: 16*(1280+512+1280)*2 = 98304 B
  logits_kernel<<<NEDGES / 16, 64, 98304, stream>>>(hb, ecb, idx, Wt_att, b_att, wbuf, hmax);

  expsum_kernel<<<(NEDGES + 255) / 256, 256, 0, stream>>>(wbuf, hmax, hsum);
  attwrite_kernel<<<(NEDGES * NHEADS + 255) / 256, 256, 0, stream>>>(wbuf, hsum, out + ATT_OFF);

  // dynamic LDS: 32*(1280+512+1280)*2 = 196608 B (WGP has 320 KB)
  gemm_pool_kernel<<<NEDGES / 32, 256, 196608, stream>>>(
      hb, ecb, idx, Wt_node, b_node, out + ATT_OFF, gf);

  gemv_kernel<<<(HIDDEN + 255) / 256, 256, 0, stream>>>(gf, W1, b1, t1, ATTEMB, HIDDEN, 1);
  gemv_kernel<<<(ATTEMB + 255) / 256, 256, 0, stream>>>(t1, W2, b2, t2, HIDDEN, ATTEMB, 0);
  gemv_kernel<<<(ATTEMB + 255) / 256, 256, 0, stream>>>(t2, W_h, nullptr, out + GF_OFF,
                                                        ATTEMB, ATTEMB, 0);
}
